// TgtMskSASent_27754078667135
// MI455X (gfx1250) — matrix operand order repaired and checked by reading, not compiled, not run on hardware
//
#include <hip/hip_runtime.h>
#include <hip/hip_bf16.h>
#include <math.h>

// ---------------------------------------------------------------------------
// Types for CDNA5 WMMA (wave32)
// ---------------------------------------------------------------------------
typedef __attribute__((ext_vector_type(16))) __bf16 bf16x16;
typedef __attribute__((ext_vector_type(8)))  __bf16 bf16x8;
typedef __attribute__((ext_vector_type(8)))  float  f32x8;
typedef __attribute__((ext_vector_type(4)))  int    i32x4;

#if __has_builtin(__builtin_amdgcn_global_load_async_to_lds_b128) && \
    __has_builtin(__builtin_amdgcn_s_wait_asynccnt)
#define USE_ASYNC_LDS 1
#else
#define USE_ASYNC_LDS 0
#endif

#if USE_ASYNC_LDS
#define AS_GLOBAL __attribute__((address_space(1)))
#define AS_LDS    __attribute__((address_space(3)))
static __device__ __forceinline__ void async_copy16(const unsigned short* g,
                                                    unsigned short* l) {
    __builtin_amdgcn_global_load_async_to_lds_b128(
        (AS_GLOBAL i32x4*)g, (AS_LDS i32x4*)l, 0, 0);
}
#endif

static __device__ __forceinline__ unsigned short f2bf(float f) {
    unsigned u = __float_as_uint(f);
    unsigned r = (u + 0x7FFFu + ((u >> 16) & 1u)) >> 16;
    return (unsigned short)r;
}
static __device__ __forceinline__ float bf2f(unsigned short h) {
    return __uint_as_float(((unsigned)h) << 16);
}

// A-fragment (16x32 bf16, M x K): lane m (0..15) holds row m, K 0..15;
// lane m+16 holds row m, K 16..31. Two 16-byte loads per lane.
static __device__ __forceinline__ bf16x16 load_a_frag(const unsigned short* p0,
                                                      const unsigned short* p1) {
    bf16x8 lo = *(const bf16x8*)p0;
    bf16x8 hi = *(const bf16x8*)p1;
    bf16x16 a;
#pragma unroll
    for (int i = 0; i < 8; ++i) { a[i] = lo[i]; a[i + 8] = hi[i]; }
    return a;
}

static __device__ __forceinline__ f32x8 wmma_bf16(bf16x16 a, bf16x16 b, f32x8 c) {
    return __builtin_amdgcn_wmma_f32_16x16x32_bf16(false, a, false, b, (short)0, c,
                                                   false, false);
}

static __device__ __forceinline__ float sigm(float x) {
    return 1.0f / (1.0f + __expf(-x));
}
static __device__ __forceinline__ float lse2(float x, float y) {
    float m = fmaxf(x, y);
    return m + logf(expf(x - m) + expf(y - m));
}

// ---------------------------------------------------------------------------
// Problem constants
// ---------------------------------------------------------------------------
#define BB 64
#define LL 512
#define EE 300
#define MDD 50
#define HH 256
#define G3H 768      // 3*H
#define K0 352       // E+MD padded to multiple of 32
#define K1 512       // layer-1 input width
#define ROWS (BB * LL)   // 32768

// ---------------------------------------------------------------------------
// 1) Embedding gather + concat -> bf16 feats (ROWS x K0), 2 zero pad cols
// ---------------------------------------------------------------------------
__global__ __launch_bounds__(K0) void embed_kernel(
    const int* __restrict__ sents, const int* __restrict__ masks,
    const float* __restrict__ wemb, const float* __restrict__ memb,
    unsigned short* __restrict__ feats) {
    int row = blockIdx.x;
    int tid = threadIdx.x;
    int w = sents[row];
    int m = masks[row];
    float v;
    if (tid < EE)            v = wemb[(size_t)w * EE + tid];
    else if (tid < EE + MDD) v = memb[m * MDD + (tid - EE)];
    else                     v = 0.0f;
    feats[(size_t)row * K0 + tid] = f2bf(v);
}

// ---------------------------------------------------------------------------
// 2) Pack f32 weight (N x K) -> bf16 (N x Kpad), zero padded
// ---------------------------------------------------------------------------
__global__ void pack_kernel(const float* __restrict__ src,
                            unsigned short* __restrict__ dst,
                            int N, int K, int Kpad) {
    int total = N * Kpad;
    for (int idx = blockIdx.x * blockDim.x + threadIdx.x; idx < total;
         idx += gridDim.x * blockDim.x) {
        int n = idx / Kpad, k = idx - n * Kpad;
        dst[idx] = (k < K) ? f2bf(src[n * K + k]) : (unsigned short)0;
    }
}

// ---------------------------------------------------------------------------
// 3) WMMA GEMM: out(MxN bf16) = A(M x K bf16, lda) * W(N x K bf16)^T + bias
//    Block = 256 thr (8 waves); block tile 64x128; wave tile 16x64.
//    A 64x32 chunk is staged into LDS with double-buffered async copies
//    (ASYNCcnt path); wave pairs (w, w+4) share the staged A rows.
//    grid = (M/64, N/128)
// ---------------------------------------------------------------------------
#define AT_STRIDE 40   // ushorts/row: 80B, 16B-aligned, bank-skewed

__global__ __launch_bounds__(256) void gemm_bias_bf16(
    const unsigned short* __restrict__ A, int lda,
    const unsigned short* __restrict__ W, int ldw,
    const float* __restrict__ bias,
    unsigned short* __restrict__ out, int N, int K) {
    __shared__ unsigned short aTile[2][64 * AT_STRIDE];

    int tid   = threadIdx.x;
    int lane  = tid & 31;
    int wid   = tid >> 5;
    int l16   = lane & 15;
    int khalf = lane >> 4;
    int mBlk  = blockIdx.x * 64;
    int n0    = blockIdx.y * 128 + (wid >> 2) * 64;

    // staging map: thread -> (row, 16B segment) of the 64x32 A chunk
    int srow = tid >> 2, sseg = tid & 3;
    const unsigned short* agbase = A + (size_t)(mBlk + srow) * lda + sseg * 8;
    unsigned short* ald0 = &aTile[0][srow * AT_STRIDE + sseg * 8];
    unsigned short* ald1 = &aTile[1][srow * AT_STRIDE + sseg * 8];

#if USE_ASYNC_LDS
    async_copy16(agbase, ald0);
#else
    *(bf16x8*)ald0 = *(const bf16x8*)agbase;
#endif

    f32x8 acc[4];
#pragma unroll
    for (int nt = 0; nt < 4; ++nt)
#pragma unroll
        for (int i = 0; i < 8; ++i) acc[nt][i] = 0.0f;

    int arowoff = ((wid & 3) * 16 + l16) * AT_STRIDE;
    int kcN = K >> 5;
    for (int kc = 0; kc < kcN; ++kc) {
        int cur = kc & 1;
#if USE_ASYNC_LDS
        __builtin_amdgcn_s_wait_asynccnt(0);  // my staged chunk has landed
#endif
        __syncthreads();                      // everyone's chunk has landed
        if (kc + 1 < kcN) {
            const unsigned short* gsrc = agbase + (kc + 1) * 32;
            unsigned short* ldst = cur ? ald0 : ald1;
#if USE_ASYNC_LDS
            async_copy16(gsrc, ldst);         // overlap with this chunk's WMMAs
#else
            *(bf16x8*)ldst = *(const bf16x8*)gsrc;
#endif
        }
        const unsigned short* arow = &aTile[cur][arowoff];
        bf16x16 a = load_a_frag(arow + khalf * 16, arow + khalf * 16 + 8);
#pragma unroll
        for (int nt = 0; nt < 4; ++nt) {
            const unsigned short* wrow =
                W + (size_t)(n0 + nt * 16 + l16) * ldw + kc * 32 + khalf * 16;
            __builtin_prefetch(wrow + 32, 0, 1);
            bf16x16 b = *(const bf16x16*)wrow;
            acc[nt] = wmma_bf16(a, b, acc[nt]);
        }
    }
#pragma unroll
    for (int nt = 0; nt < 4; ++nt) {
        int col = n0 + nt * 16 + l16;
        float bv = bias[col];
#pragma unroll
        for (int i = 0; i < 8; ++i) {
            int row = mBlk + (wid & 3) * 16 + i + khalf * 8;
            out[(size_t)row * N + col] = f2bf(acc[nt][i] + bv);
        }
    }
}

// ---------------------------------------------------------------------------
// 4) GRU recurrence (one layer, both directions).
//    grid = 8 blocks: dir = b>>2 (0=fwd,1=bwd), mq = b&3 (16 batch rows).
//    256 threads = 8 waves. Per step: gh(16x768) = h(16x256, LDS bf16) @
//    W_hh^T via WMMA (W streamed from L2), then fused gate update.
// ---------------------------------------------------------------------------
#define HS_STRIDE 264   // 16B-aligned, 4-bank skew per row
#define GH_STRIDE 776

__global__ __launch_bounds__(256) void gru_recur(
    const unsigned short* __restrict__ gxF, const unsigned short* __restrict__ gxB,
    const unsigned short* __restrict__ whhF, const unsigned short* __restrict__ whhB,
    const float* __restrict__ bhhF, const float* __restrict__ bhhB,
    const int* __restrict__ lens,
    unsigned short* __restrict__ outBf, float* __restrict__ outF32,
    int layer1) {
    __shared__ unsigned short hstate[16 * HS_STRIDE];
    __shared__ float ghbuf[16 * GH_STRIDE];

    int dir = blockIdx.x >> 2;
    int mq  = blockIdx.x & 3;
    const unsigned short* gx  = dir ? gxB  : gxF;
    const unsigned short* whh = dir ? whhB : whhF;
    const float*          bhh = dir ? bhhB : bhhF;

    int tid   = threadIdx.x;
    int lane  = tid & 31;
    int wid   = tid >> 5;
    int l16   = lane & 15;
    int khalf = lane >> 4;

    int row_e = tid & 15;
    int jbase = (tid >> 4) * 16;
    int b_e   = mq * 16 + row_e;
    int len_e = lens[b_e];

    for (int i = tid; i < 16 * HS_STRIDE; i += 256) hstate[i] = 0;
    __syncthreads();

    for (int t = 0; t < LL; ++t) {
        f32x8 acc[6];
#pragma unroll
        for (int nt = 0; nt < 6; ++nt)
#pragma unroll
            for (int i = 0; i < 8; ++i) acc[nt][i] = 0.0f;

        for (int k = 0; k < 8; ++k) {
            const unsigned short* hrow = &hstate[l16 * HS_STRIDE + k * 32];
            bf16x16 a = load_a_frag(hrow + khalf * 16, hrow + khalf * 16 + 8);
#pragma unroll
            for (int nt = 0; nt < 6; ++nt) {
                int wcol = wid * 96 + nt * 16 + l16;
                const unsigned short* wrow =
                    whh + (size_t)wcol * HH + k * 32 + khalf * 16;
                __builtin_prefetch(wrow + 32, 0, 1);
                bf16x16 b = *(const bf16x16*)wrow;
                acc[nt] = wmma_bf16(a, b, acc[nt]);
            }
        }
#pragma unroll
        for (int nt = 0; nt < 6; ++nt) {
            int col = wid * 96 + nt * 16 + l16;
#pragma unroll
            for (int i = 0; i < 8; ++i)
                ghbuf[(i + khalf * 8) * GH_STRIDE + col] = acc[nt][i];
        }
        __syncthreads();

        int tg = t;
        if (dir) tg = (t < len_e) ? (len_e - 1 - t) : t;
        const unsigned short* gxr = gx + ((size_t)b_e * LL + tg) * G3H;
        bool mvalid = (t < len_e);
        int tout = dir ? tg : t;

#pragma unroll 4
        for (int jj = 0; jj < 16; ++jj) {
            int j = jbase + jj;
            float ghr = ghbuf[row_e * GH_STRIDE + j]        + bhh[j];
            float ghz = ghbuf[row_e * GH_STRIDE + HH + j]   + bhh[HH + j];
            float ghn = ghbuf[row_e * GH_STRIDE + 2*HH + j] + bhh[2*HH + j];
            float r = sigm(bf2f(gxr[j]) + ghr);
            float z = sigm(bf2f(gxr[HH + j]) + ghz);
            float n = tanhf(bf2f(gxr[2*HH + j]) + r * ghn);
            float hp = bf2f(hstate[row_e * HS_STRIDE + j]);
            float hn_ = (1.0f - z) * n + z * hp;
            float h  = mvalid ? hn_ : hp;
            float ov = mvalid ? h : 0.0f;
            size_t oidx = ((size_t)b_e * LL + tout) * 512 + dir * HH + j;
            if (layer1) outF32[oidx] = ov;
            else        outBf[oidx]  = f2bf(ov);
            hstate[row_e * HS_STRIDE + j] = f2bf(h);
        }
        __syncthreads();
    }
}

// ---------------------------------------------------------------------------
// 5) tri = context @ w_tri^T + b_tri  (ROWS x 2)
// ---------------------------------------------------------------------------
__global__ __launch_bounds__(128) void tri_kernel(
    const float* __restrict__ ctx, const float* __restrict__ w_tri,
    const float* __restrict__ b_tri, float* __restrict__ tri) {
    int row = blockIdx.x;
    int tid = threadIdx.x;
    const float* c = ctx + (size_t)row * 512;
    float a0 = 0.0f, a1 = 0.0f;
    for (int d = tid; d < 512; d += 128) {
        float v = c[d];
        a0 += v * w_tri[d];
        a1 += v * w_tri[512 + d];
    }
#pragma unroll
    for (int off = 16; off > 0; off >>= 1) {
        a0 += __shfl_xor(a0, off, 32);
        a1 += __shfl_xor(a1, off, 32);
    }
    __shared__ float s0[4], s1[4];
    if ((tid & 31) == 0) { s0[tid >> 5] = a0; s1[tid >> 5] = a1; }
    __syncthreads();
    if (tid == 0) {
        tri[(size_t)row * 2 + 0] = s0[0] + s0[1] + s0[2] + s0[3] + b_tri[0];
        tri[(size_t)row * 2 + 1] = s1[0] + s1[1] + s1[2] + s1[3] + b_tri[1];
    }
}

// ---------------------------------------------------------------------------
// 6) CRF forward/backward marginals (2 states): one lane per batch element
// ---------------------------------------------------------------------------
__global__ __launch_bounds__(64) void crf_kernel(
    const float* __restrict__ tri, const int* __restrict__ lens,
    const float* __restrict__ T, float* __restrict__ alphas,
    float* __restrict__ sel, float* __restrict__ selsum) {
    int b = threadIdx.x;
    if (b >= BB) return;
    float T00 = T[0], T01 = T[1], T10 = T[2], T11 = T[3];
    int len = lens[b];
    const float* e = tri + (size_t)b * LL * 2;
    float* al = alphas + (size_t)b * LL * 2;

    float a0 = e[0], a1 = e[1];
    al[0] = a0; al[1] = a1;
    for (int t = 1; t < LL; ++t) {
        if (t < len) {
            float n0 = e[t * 2 + 0] + lse2(a0 + T00, a1 + T10);
            float n1 = e[t * 2 + 1] + lse2(a0 + T01, a1 + T11);
            a0 = n0; a1 = n1;
        }
        al[t * 2] = a0; al[t * 2 + 1] = a1;
    }
    float logZ = lse2(a0, a1);

    float b0 = 0.0f, b1 = 0.0f, ssum = 0.0f;
    {
        float m_t = (LL - 1 < len) ? 1.0f : 0.0f;
        float s = expf(al[(LL - 1) * 2 + 1] - logZ) * m_t;
        sel[(size_t)b * LL + LL - 1] = s; ssum += s;
    }
    for (int t = LL - 2; t >= 0; --t) {
        float m_n = (t + 1 < len) ? 1.0f : 0.0f;
        float n0 = lse2(T00 + e[(t + 1) * 2 + 0] + b0, T01 + e[(t + 1) * 2 + 1] + b1);
        float n1 = lse2(T10 + e[(t + 1) * 2 + 0] + b0, T11 + e[(t + 1) * 2 + 1] + b1);
        b0 = m_n ? n0 : 0.0f; b1 = m_n ? n1 : 0.0f;
        float m_t = (t < len) ? 1.0f : 0.0f;
        float s = expf(al[t * 2 + 1] + b1 - logZ) * m_t;
        sel[(size_t)b * LL + t] = s; ssum += s;
    }
    selsum[b] = ssum;
}

// ---------------------------------------------------------------------------
// 7) sent_v[b][d] = sum_l sel[b][l] * ctx[b][l][d]
// ---------------------------------------------------------------------------
__global__ __launch_bounds__(512) void sentv_kernel(
    const float* __restrict__ sel, const float* __restrict__ ctx,
    float* __restrict__ sentv) {
    int b = blockIdx.x, d = threadIdx.x;
    const float* c = ctx + (size_t)b * LL * 512;
    const float* s = sel + (size_t)b * LL;
    float acc = 0.0f;
    for (int l = 0; l < LL; ++l) acc += s[l] * c[(size_t)l * 512 + d];
    sentv[(size_t)b * 512 + d] = acc;
}

// ---------------------------------------------------------------------------
// 8) scores + loss + penalty -> d_out[2]
// ---------------------------------------------------------------------------
__global__ __launch_bounds__(64) void final_kernel(
    const float* __restrict__ sentv, const float* __restrict__ w_label,
    const float* __restrict__ b_label, const int* __restrict__ labels,
    const float* __restrict__ T, const float* __restrict__ selsum,
    float* __restrict__ out) {
    __shared__ float rloss[64], rsum[64];
    int b = threadIdx.x;
    float sc[3];
#pragma unroll
    for (int c = 0; c < 3; ++c) {
        float a = b_label[c];
        const float* w = w_label + c * 512;
        const float* v = sentv + (size_t)b * 512;
        for (int d = 0; d < 512; ++d) a += w[d] * v[d];
        sc[c] = a;
    }
    float mx = fmaxf(sc[0], fmaxf(sc[1], sc[2]));
    float lz = mx + logf(expf(sc[0] - mx) + expf(sc[1] - mx) + expf(sc[2] - mx));
    rloss[b] = -(sc[labels[b]] - lz);
    rsum[b] = selsum[b];
    __syncthreads();
    if (b == 0) {
        float sl = 0.0f, ss = 0.0f;
        for (int i = 0; i < BB; ++i) { sl += rloss[i]; ss += rsum[i]; }
        float pena = fmaxf(T[2] - T[0], 0.0f) + fmaxf(T[1] - T[3], 0.0f);
        out[0] = sl / (float)BB;
        out[1] = 0.1f * pena + 0.01f * (ss / (float)BB);
    }
}

// ---------------------------------------------------------------------------
// Host launcher
// ---------------------------------------------------------------------------
extern "C" void kernel_launch(void* const* d_in, const int* in_sizes, int n_in,
                              void* d_out, int out_size, void* d_ws, size_t ws_size,
                              hipStream_t stream) {
    (void)in_sizes; (void)n_in; (void)out_size; (void)ws_size;
    const int*   sents = (const int*)d_in[0];
    const int*   masks = (const int*)d_in[1];
    const int*   labels = (const int*)d_in[2];
    const int*   lens  = (const int*)d_in[3];
    const float* wemb  = (const float*)d_in[4];
    const float* memb  = (const float*)d_in[5];
    // gru dict in insertion order: (w_ih, w_hh, b_ih, b_hh) x {l0f,l0b,l1f,l1b}
    const float* w_ih_l0_f = (const float*)d_in[6];
    const float* w_hh_l0_f = (const float*)d_in[7];
    const float* b_ih_l0_f = (const float*)d_in[8];
    const float* b_hh_l0_f = (const float*)d_in[9];
    const float* w_ih_l0_b = (const float*)d_in[10];
    const float* w_hh_l0_b = (const float*)d_in[11];
    const float* b_ih_l0_b = (const float*)d_in[12];
    const float* b_hh_l0_b = (const float*)d_in[13];
    const float* w_ih_l1_f = (const float*)d_in[14];
    const float* w_hh_l1_f = (const float*)d_in[15];
    const float* b_ih_l1_f = (const float*)d_in[16];
    const float* b_hh_l1_f = (const float*)d_in[17];
    const float* w_ih_l1_b = (const float*)d_in[18];
    const float* w_hh_l1_b = (const float*)d_in[19];
    const float* b_ih_l1_b = (const float*)d_in[20];
    const float* b_hh_l1_b = (const float*)d_in[21];
    const float* w_tri = (const float*)d_in[22];
    const float* b_tri = (const float*)d_in[23];
    const float* trans = (const float*)d_in[24];
    const float* w_label = (const float*)d_in[25];
    const float* b_label = (const float*)d_in[26];

    char* ws = (char*)d_ws;
    size_t off = 0;
    auto take = [&](size_t bytes) -> char* {
        char* p = ws + off;
        off = (off + bytes + 255) & ~(size_t)255;
        return p;
    };
    unsigned short* feats = (unsigned short*)take((size_t)ROWS * K0 * 2);
    unsigned short* w0f   = (unsigned short*)take((size_t)G3H * K0 * 2);
    unsigned short* w0b   = (unsigned short*)take((size_t)G3H * K0 * 2);
    unsigned short* whh0f = (unsigned short*)take((size_t)G3H * HH * 2);
    unsigned short* whh0b = (unsigned short*)take((size_t)G3H * HH * 2);
    unsigned short* whh1f = (unsigned short*)take((size_t)G3H * HH * 2);
    unsigned short* whh1b = (unsigned short*)take((size_t)G3H * HH * 2);
    unsigned short* w1f   = (unsigned short*)take((size_t)G3H * K1 * 2);
    unsigned short* w1b   = (unsigned short*)take((size_t)G3H * K1 * 2);
    unsigned short* gxF   = (unsigned short*)take((size_t)ROWS * G3H * 2);
    unsigned short* gxB   = (unsigned short*)take((size_t)ROWS * G3H * 2);
    unsigned short* h0    = (unsigned short*)take((size_t)ROWS * 512 * 2);
    float* ctx    = (float*)take((size_t)ROWS * 512 * 4);
    float* tri    = (float*)take((size_t)ROWS * 2 * 4);
    float* alphas = (float*)take((size_t)ROWS * 2 * 4);
    float* sel    = (float*)take((size_t)ROWS * 4);
    float* selsum = (float*)take((size_t)BB * 4);
    float* sentv  = (float*)take((size_t)BB * 512 * 4);

    embed_kernel<<<ROWS, K0, 0, stream>>>(sents, masks, wemb, memb, feats);

    pack_kernel<<<256, 256, 0, stream>>>(w_ih_l0_f, w0f, G3H, EE + MDD, K0);
    pack_kernel<<<256, 256, 0, stream>>>(w_ih_l0_b, w0b, G3H, EE + MDD, K0);
    pack_kernel<<<256, 256, 0, stream>>>(w_hh_l0_f, whh0f, G3H, HH, HH);
    pack_kernel<<<256, 256, 0, stream>>>(w_hh_l0_b, whh0b, G3H, HH, HH);
    pack_kernel<<<256, 256, 0, stream>>>(w_hh_l1_f, whh1f, G3H, HH, HH);
    pack_kernel<<<256, 256, 0, stream>>>(w_hh_l1_b, whh1b, G3H, HH, HH);
    pack_kernel<<<256, 256, 0, stream>>>(w_ih_l1_f, w1f, G3H, K1, K1);
    pack_kernel<<<256, 256, 0, stream>>>(w_ih_l1_b, w1b, G3H, K1, K1);

    dim3 gg(ROWS / 64, G3H / 128);
    gemm_bias_bf16<<<gg, 256, 0, stream>>>(feats, K0, w0f, K0, b_ih_l0_f, gxF, G3H, K0);
    gemm_bias_bf16<<<gg, 256, 0, stream>>>(feats, K0, w0b, K0, b_ih_l0_b, gxB, G3H, K0);

    gru_recur<<<8, 256, 0, stream>>>(gxF, gxB, whh0f, whh0b, b_hh_l0_f, b_hh_l0_b,
                                     lens, h0, ctx, 0);

    gemm_bias_bf16<<<gg, 256, 0, stream>>>(h0, K1, w1f, K1, b_ih_l1_f, gxF, G3H, K1);
    gemm_bias_bf16<<<gg, 256, 0, stream>>>(h0, K1, w1b, K1, b_ih_l1_b, gxB, G3H, K1);

    gru_recur<<<8, 256, 0, stream>>>(gxF, gxB, whh1f, whh1b, b_hh_l1_f, b_hh_l1_b,
                                     lens, h0, ctx, 1);

    tri_kernel<<<ROWS, 128, 0, stream>>>(ctx, w_tri, b_tri, tri);
    crf_kernel<<<1, 64, 0, stream>>>(tri, lens, trans, alphas, sel, selsum);
    sentv_kernel<<<BB, 512, 0, stream>>>(sel, ctx, sentv);
    final_kernel<<<1, 64, 0, stream>>>(sentv, w_label, b_label, labels, trans,
                                       selsum, (float*)d_out);
}
